// Layer_60249801228620
// MI455X (gfx1250) — matrix operand order repaired and checked by reading, not compiled, not run on hardware
//
#include <hip/hip_runtime.h>
#include <hip/hip_bf16.h>
#include <cmath>
#include <cstdint>

// ---------------------------------------------------------------------------
// MI455X / gfx1250 implementation.
// Pipeline (all GEMMs: C[M,N] = A[M,K] * B[N,K]^T, bf16 in, f32 accum WMMA):
//   1. convert weights -> bf16; expw = bf16(exp(pos_bias))
//   2. layernorm(data) -> h (bf16)           [shared by both branches]
//   3. q GEMM  -> sq = sigmoid(q)  (f32)
//      k GEMM  -> ekT[(b*D+d)*T + t] = bf16(exp(k))      (transposed store)
//      v GEMM  -> ekvT[idx] = bf16(ekT[idx] * v)         (transposed store)
//   4. num = expw @ ekvT^T  (f32), den = expw @ ekT^T (f32)
//   5. s = bf16(sq * num/den)
//   6. aft = s @ out_w^T + out_b (f32)
//   7. x12 = h @ fc1_w^T + fc1_b (f32); g = bf16(x1 * mish(gate))
//   8. out = data + aft + g @ fc2_w^T + fc2_b
// ---------------------------------------------------------------------------

typedef __attribute__((ext_vector_type(16))) __bf16 v16bf;
typedef __attribute__((ext_vector_type(8)))  __bf16 v8bf;
typedef __attribute__((ext_vector_type(8)))  float  v8f;
typedef __attribute__((ext_vector_type(4)))  int    v4i;

#define TPB 256
constexpr int LDSS = 40;   // LDS row stride (elements): 80B rows -> conflict-free frag loads

enum { EPI_RAW = 0, EPI_BIAS = 1, EPI_SIGMOID = 2, EPI_K_EXPT = 3, EPI_V_EKV = 4, EPI_FINAL = 5 };

// ---- async global->LDS staging (CDNA5 ASYNCcnt path), with fallback --------
#if __has_builtin(__builtin_amdgcn_global_load_async_to_lds_b128)
#define HAS_ASYNC_LDS 1
typedef __attribute__((address_space(1))) v4i* gptr_v4i;
typedef __attribute__((address_space(3))) v4i* lptr_v4i;
__device__ __forceinline__ void stage16_async(const __bf16* g, __bf16* l)
{
  __builtin_amdgcn_global_load_async_to_lds_b128(
      (gptr_v4i)(uintptr_t)g,
      (lptr_v4i)(uintptr_t)l,
      0, 0);
}
__device__ __forceinline__ void wait_async()
{
#if __has_builtin(__builtin_amdgcn_s_wait_asynccnt)
  __builtin_amdgcn_s_wait_asynccnt(0);
#else
  asm volatile("s_wait_asynccnt 0x0" ::: "memory");
#endif
}
#else
#define HAS_ASYNC_LDS 0
#endif

// ---------------------------------------------------------------------------
// Tiled WMMA GEMM: 128x128 block tile, 8 waves (2x4), 64x32 per wave,
// K-step 32 (one v_wmma_f32_16x16x32_bf16 per 16x16 tile per step),
// double-buffered LDS staging (async-to-LDS when available).
// ---------------------------------------------------------------------------
template<int EPI>
__launch_bounds__(TPB)
__global__ void gemm_bf16_kernel(const __bf16* __restrict__ A,
                                 const __bf16* __restrict__ Bw,
                                 const float*  __restrict__ bias,
                                 float*        __restrict__ outF,
                                 __bf16*       __restrict__ outB,
                                 const __bf16* __restrict__ ekT,
                                 const float*  __restrict__ add1,
                                 const float*  __restrict__ add2,
                                 int N, int K)
{
  __shared__ __bf16 As[2][128 * LDSS];
  __shared__ __bf16 Bs[2][128 * LDSS];

  const int tid  = threadIdx.x;
  const int lane = tid & 31;
  const int w    = tid >> 5;
  const int wm   = w >> 2;        // 0..1 : wave row in 2x4 wave grid
  const int wn   = w & 3;         // 0..3 : wave col
  const int lm   = lane & 15;
  const int lh   = lane >> 4;

  const size_t m0 = (size_t)blockIdx.y * 128;
  const size_t n0 = (size_t)blockIdx.x * 128;

  // staging: 512 16B chunks per tile, 2 per thread
  const int sr = tid >> 2;            // row 0..63 (and +64)
  const int sk = (tid & 3) * 8;       // k element offset 0/8/16/24

  const __bf16* Abase = A  + m0 * (size_t)K + sk;
  const __bf16* Bbase = Bw + n0 * (size_t)K + sk;

  v8f acc[4][2] = {};

  // fragment loads + 8 WMMAs for one 32-deep K step on buffer (ab, bb)
  auto compute_step = [&](const __bf16* ab, const __bf16* bb) {
    v16bf af[4];
#pragma unroll
    for (int i = 0; i < 4; ++i) {
      const __bf16* p = ab + (wm * 64 + i * 16 + lm) * LDSS + lh * 8;
      v8bf lo = *(const v8bf*)p;          // K = lh*8 .. lh*8+7
      v8bf hi = *(const v8bf*)(p + 16);   // K = 16+lh*8 ..
      af[i] = __builtin_shufflevector(lo, hi, 0,1,2,3,4,5,6,7,8,9,10,11,12,13,14,15);
    }
    v16bf bfr[2];
#pragma unroll
    for (int j = 0; j < 2; ++j) {
      const __bf16* p = bb + (wn * 32 + j * 16 + lm) * LDSS + lh * 8;
      v8bf lo = *(const v8bf*)p;
      v8bf hi = *(const v8bf*)(p + 16);
      bfr[j] = __builtin_shufflevector(lo, hi, 0,1,2,3,4,5,6,7,8,9,10,11,12,13,14,15);
    }
#pragma unroll
    for (int i = 0; i < 4; ++i)
#pragma unroll
      for (int j = 0; j < 2; ++j)
        acc[i][j] = __builtin_amdgcn_wmma_f32_16x16x32_bf16(
            false, af[i], false, bfr[j], (short)0, acc[i][j], false, false);
  };

#if HAS_ASYNC_LDS
  // ---- async-to-LDS double-buffered pipeline (no staging VGPRs) ----
  stage16_async(Abase + (size_t)sr        * K, &As[0][sr * LDSS + sk]);
  stage16_async(Abase + (size_t)(sr + 64) * K, &As[0][(sr + 64) * LDSS + sk]);
  stage16_async(Bbase + (size_t)sr        * K, &Bs[0][sr * LDSS + sk]);
  stage16_async(Bbase + (size_t)(sr + 64) * K, &Bs[0][(sr + 64) * LDSS + sk]);
  wait_async();
  __syncthreads();

  int buf = 0;
  for (int k0 = 0; k0 < K; k0 += 32) {
    const bool nxt = (k0 + 32) < K;
    if (nxt) {  // issue async copies into the other buffer; overlaps the WMMAs
      const int nb = buf ^ 1;
      const __bf16* Ap = Abase + (k0 + 32);
      const __bf16* Bp = Bbase + (k0 + 32);
      stage16_async(Ap + (size_t)sr        * K, &As[nb][sr * LDSS + sk]);
      stage16_async(Ap + (size_t)(sr + 64) * K, &As[nb][(sr + 64) * LDSS + sk]);
      stage16_async(Bp + (size_t)sr        * K, &Bs[nb][sr * LDSS + sk]);
      stage16_async(Bp + (size_t)(sr + 64) * K, &Bs[nb][(sr + 64) * LDSS + sk]);
    }
    compute_step(As[buf], Bs[buf]);
    if (nxt) {
      wait_async();
      __syncthreads();
      buf ^= 1;
    }
  }
#else
  // ---- fallback: register-prefetch pipeline (load -> VGPR -> ds_store) ----
  {
    uint4 a0 = *(const uint4*)(Abase + (size_t)sr        * K);
    uint4 a1 = *(const uint4*)(Abase + (size_t)(sr + 64) * K);
    uint4 b0 = *(const uint4*)(Bbase + (size_t)sr        * K);
    uint4 b1 = *(const uint4*)(Bbase + (size_t)(sr + 64) * K);
    *(uint4*)&As[0][sr * LDSS + sk]        = a0;
    *(uint4*)&As[0][(sr + 64) * LDSS + sk] = a1;
    *(uint4*)&Bs[0][sr * LDSS + sk]        = b0;
    *(uint4*)&Bs[0][(sr + 64) * LDSS + sk] = b1;
  }
  __syncthreads();

  int buf = 0;
  for (int k0 = 0; k0 < K; k0 += 32) {
    uint4 a0, a1, b0, b1;
    const bool nxt = (k0 + 32) < K;
    if (nxt) {
      const __bf16* Ap = Abase + (k0 + 32);
      const __bf16* Bp = Bbase + (k0 + 32);
      a0 = *(const uint4*)(Ap + (size_t)sr        * K);
      a1 = *(const uint4*)(Ap + (size_t)(sr + 64) * K);
      b0 = *(const uint4*)(Bp + (size_t)sr        * K);
      b1 = *(const uint4*)(Bp + (size_t)(sr + 64) * K);
    }
    compute_step(As[buf], Bs[buf]);
    if (nxt) {
      const int nb = buf ^ 1;
      *(uint4*)&As[nb][sr * LDSS + sk]        = a0;
      *(uint4*)&As[nb][(sr + 64) * LDSS + sk] = a1;
      *(uint4*)&Bs[nb][sr * LDSS + sk]        = b0;
      *(uint4*)&Bs[nb][(sr + 64) * LDSS + sk] = b1;
      __syncthreads();
      buf = nb;
    }
  }
#endif

  // ---- epilogue ----
#pragma unroll
  for (int i = 0; i < 4; ++i) {
#pragma unroll
    for (int j = 0; j < 2; ++j) {
      const size_t ng = n0 + wn * 32 + j * 16 + lm;
      const size_t mb = m0 + wm * 64 + i * 16 + lh * 8;
      float bv = 0.f;
      if constexpr (EPI != EPI_RAW) bv = bias[ng];
#pragma unroll
      for (int r = 0; r < 8; ++r) {
        const size_t mg = mb + r;
        const float v = acc[i][j][r] + bv;
        if constexpr (EPI == EPI_RAW || EPI == EPI_BIAS) {
          outF[mg * (size_t)N + ng] = v;
        } else if constexpr (EPI == EPI_SIGMOID) {
          outF[mg * (size_t)N + ng] = 1.f / (1.f + __expf(-v));
        } else if constexpr (EPI == EPI_K_EXPT) {
          const size_t t = mg >> 3, b = mg & 7;          // row = t*8+b
          outB[(b * 1024 + ng) * 2048 + t] = (__bf16)__expf(v);
        } else if constexpr (EPI == EPI_V_EKV) {
          const size_t t = mg >> 3, b = mg & 7;
          const size_t idx = (b * 1024 + ng) * 2048 + t;
          outB[idx] = (__bf16)((float)ekT[idx] * v);
        } else if constexpr (EPI == EPI_FINAL) {
          const size_t idx = mg * (size_t)N + ng;
          outF[idx] = add1[idx] + add2[idx] + v;
        }
      }
    }
  }
}

// --------------------------- elementwise kernels ---------------------------

__global__ void cvt_kernel(const float* __restrict__ in, __bf16* __restrict__ out, int doexp)
{
  const size_t i = (size_t)blockIdx.x * TPB + threadIdx.x;
  float v = in[i];
  if (doexp) v = __expf(v);
  out[i] = (__bf16)v;
}

__global__ void ln_kernel(const float* __restrict__ x, __bf16* __restrict__ h)
{
  const size_t row = blockIdx.x;
  const float* xr = x + row * 1024;
  float s = 0.f, ss = 0.f;
  for (int i = threadIdx.x; i < 1024; i += TPB) {
    const float v = xr[i];
    s += v; ss += v * v;
  }
  for (int o = 16; o; o >>= 1) { s += __shfl_xor(s, o); ss += __shfl_xor(ss, o); }
  __shared__ float ps[8], pss[8];
  const int wv = threadIdx.x >> 5;
  if ((threadIdx.x & 31) == 0) { ps[wv] = s; pss[wv] = ss; }
  __syncthreads();
  float S = 0.f, SS = 0.f;
#pragma unroll
  for (int i = 0; i < 8; ++i) { S += ps[i]; SS += pss[i]; }
  const float m   = S * (1.f / 1024.f);
  const float var = SS * (1.f / 1024.f) - m * m;
  const float inv = rsqrtf(var + 1.17549435e-38f);   // np.finfo(f32).tiny
  for (int i = threadIdx.x; i < 1024; i += TPB)
    h[row * 1024 + i] = (__bf16)((xr[i] - m) * inv);
}

__global__ void ratio_kernel(const float* __restrict__ sq, const float* __restrict__ num,
                             const float* __restrict__ den, __bf16* __restrict__ s)
{
  const size_t i = (size_t)blockIdx.x * TPB + threadIdx.x;
  s[i] = (__bf16)(sq[i] * (num[i] / den[i]));
}

__global__ void glu_kernel(const float* __restrict__ x12, __bf16* __restrict__ g)
{
  const size_t i = (size_t)blockIdx.x * TPB + threadIdx.x;
  const size_t r = i >> 10, d = i & 1023;
  const float x1 = x12[r * 2048 + d];
  const float gt = x12[r * 2048 + 1024 + d];
  const float sp = (gt > 20.f) ? gt : log1pf(__expf(gt));   // softplus
  g[i] = (__bf16)(x1 * gt * tanhf(sp));                     // x1 * mish(gate)
}

// --------------------------------- host ------------------------------------

extern "C" void kernel_launch(void* const* d_in, const int* in_sizes, int n_in,
                              void* d_out, int out_size, void* d_ws, size_t ws_size,
                              hipStream_t stream)
{
  (void)in_sizes; (void)n_in; (void)out_size; (void)ws_size;
  const float* data  = (const float*)d_in[0];
  const float* qkv_w = (const float*)d_in[1];
  const float* qkv_b = (const float*)d_in[2];
  const float* pos_b = (const float*)d_in[3];
  const float* out_w = (const float*)d_in[4];
  const float* out_b = (const float*)d_in[5];
  const float* fc1_w = (const float*)d_in[6];
  const float* fc1_b = (const float*)d_in[7];
  const float* fc2_w = (const float*)d_in[8];
  const float* fc2_b = (const float*)d_in[9];
  float* out = (float*)d_out;

  char* ws = (char*)d_ws;
  size_t off = 0;
  auto take = [&](size_t bytes) { void* p = ws + off; off += (bytes + 255) & ~(size_t)255; return p; };

  __bf16* h_bf    = (__bf16*)take((size_t)16384 * 1024 * 2);
  __bf16* wqkv_bf = (__bf16*)take((size_t)3072  * 1024 * 2);
  __bf16* wout_bf = (__bf16*)take((size_t)1024  * 1024 * 2);
  __bf16* wfc1_bf = (__bf16*)take((size_t)2048  * 1024 * 2);
  __bf16* wfc2_bf = (__bf16*)take((size_t)1024  * 1024 * 2);
  __bf16* expw_bf = (__bf16*)take((size_t)2048  * 2048 * 2);
  __bf16* ekT_bf  = (__bf16*)take((size_t)2048  * 8192 * 2);
  __bf16* ekvT_bf = (__bf16*)take((size_t)2048  * 8192 * 2);
  float*  num_f   = (float*) take((size_t)2048  * 8192 * 4);
  float*  sq_f    = (float*) take((size_t)16384 * 1024 * 4);
  float*  den_f   = (float*) take((size_t)2048  * 8192 * 4);  // contiguous after sq_f
  // reuse (dependency-safe):
  __bf16* s_bf  = ekvT_bf;   // after num GEMM, ekvT is free
  float*  aft_f = num_f;     // after ratio kernel, num is free
  float*  x12_f = sq_f;      // 128MB spanning sq_f + den_f (both free after ratio)
  __bf16* g_bf  = ekT_bf;    // after den GEMM, ekT is free

  // 1. weight conversions + exp(pos_bias)
  cvt_kernel<<<dim3((3072u * 1024u) / TPB), TPB, 0, stream>>>(qkv_w, wqkv_bf, 0);
  cvt_kernel<<<dim3((1024u * 1024u) / TPB), TPB, 0, stream>>>(out_w, wout_bf, 0);
  cvt_kernel<<<dim3((2048u * 1024u) / TPB), TPB, 0, stream>>>(fc1_w, wfc1_bf, 0);
  cvt_kernel<<<dim3((1024u * 1024u) / TPB), TPB, 0, stream>>>(fc2_w, wfc2_bf, 0);
  cvt_kernel<<<dim3((2048u * 2048u) / TPB), TPB, 0, stream>>>(pos_b, expw_bf, 1);

  // 2. layernorm
  ln_kernel<<<dim3(16384), TPB, 0, stream>>>(data, h_bf);

  // 3. q / k / v projections (M=16384, N=1024, K=1024)
  gemm_bf16_kernel<EPI_SIGMOID><<<dim3(8, 128), TPB, 0, stream>>>(
      h_bf, wqkv_bf, qkv_b, sq_f, nullptr, nullptr, nullptr, nullptr, 1024, 1024);
  gemm_bf16_kernel<EPI_K_EXPT><<<dim3(8, 128), TPB, 0, stream>>>(
      h_bf, wqkv_bf + (size_t)1024 * 1024, qkv_b + 1024, nullptr, ekT_bf, nullptr, nullptr, nullptr, 1024, 1024);
  gemm_bf16_kernel<EPI_V_EKV><<<dim3(8, 128), TPB, 0, stream>>>(
      h_bf, wqkv_bf + (size_t)2048 * 1024, qkv_b + 2048, nullptr, ekvT_bf, ekT_bf, nullptr, nullptr, 1024, 1024);

  // 4. AFT num/den (M=2048, N=8192, K=2048)
  gemm_bf16_kernel<EPI_RAW><<<dim3(64, 16), TPB, 0, stream>>>(
      expw_bf, ekvT_bf, nullptr, num_f, nullptr, nullptr, nullptr, nullptr, 8192, 2048);
  gemm_bf16_kernel<EPI_RAW><<<dim3(64, 16), TPB, 0, stream>>>(
      expw_bf, ekT_bf, nullptr, den_f, nullptr, nullptr, nullptr, nullptr, 8192, 2048);

  // 5. s = sigmoid(q) * num / den
  ratio_kernel<<<dim3(65536), TPB, 0, stream>>>(sq_f, num_f, den_f, s_bf);

  // 6. aft = s @ out_w^T + out_b
  gemm_bf16_kernel<EPI_BIAS><<<dim3(8, 128), TPB, 0, stream>>>(
      s_bf, wout_bf, out_b, aft_f, nullptr, nullptr, nullptr, nullptr, 1024, 1024);

  // 7. fc1 + mish-gated GLU
  gemm_bf16_kernel<EPI_BIAS><<<dim3(16, 128), TPB, 0, stream>>>(
      h_bf, wfc1_bf, fc1_b, x12_f, nullptr, nullptr, nullptr, nullptr, 2048, 1024);
  glu_kernel<<<dim3(65536), TPB, 0, stream>>>(x12_f, g_bf);

  // 8. out = data + aft + g @ fc2_w^T + fc2_b
  gemm_bf16_kernel<EPI_FINAL><<<dim3(8, 128), TPB, 0, stream>>>(
      g_bf, wfc2_bf, fc2_b, out, nullptr, nullptr, data, aft_f, 1024, 1024);
}
